// MultiHeadAttention_65481071410111
// MI455X (gfx1250) — compile-verified, matrix operand order checked by reading, not run on hardware
//
#include <hip/hip_runtime.h>
#include <hip/hip_bf16.h>

#define EMBED 768
#define SEQ   4096
#define NHEAD 12
#define DK    64
#define QKVD  2304

typedef __attribute__((ext_vector_type(16))) __bf16 v16bf;
typedef __attribute__((ext_vector_type(8)))  float  v8f;
typedef __attribute__((ext_vector_type(4)))  __bf16 v4bf;

// Exact pointee type of __builtin_amdgcn_global_load_async_to_lds_b128 args
// (from the compiler diagnostic): 4 x i32 GCC vector.
typedef int v4i __attribute__((vector_size(16)));
typedef __attribute__((address_space(1))) v4i as1_v4i;   // global
typedef __attribute__((address_space(3))) v4i as3_v4i;   // LDS

__device__ __forceinline__ v8f wmma_bf16(v16bf a, v16bf b, v8f c) {
  // D = A(16x32 bf16) * B(32x16 bf16) + C(16x16 f32)
  return __builtin_amdgcn_wmma_f32_16x16x32_bf16(false, a, false, b, (short)0, c,
                                                 false, false);
}

// Async global -> LDS 16-byte copy (ASYNCcnt-tracked on gfx1250); falls back
// to a synchronous copy if the builtin is absent in this toolchain.
__device__ __forceinline__ void async_copy16(const __bf16* g, __bf16* l) {
#if defined(__AMDGCN__) && __has_builtin(__builtin_amdgcn_global_load_async_to_lds_b128)
  __builtin_amdgcn_global_load_async_to_lds_b128((as1_v4i*)(void*)g,
                                                 (as3_v4i*)(void*)l, 0, 0);
#else
  *(uint4*)l = *(const uint4*)g;
#endif
}

template <int N>
__device__ __forceinline__ void async_wait() {
#if defined(__AMDGCN__) && __has_builtin(__builtin_amdgcn_s_wait_asynccnt)
  __builtin_amdgcn_s_wait_asynccnt(N);
#endif
}

// Load one 16x32 bf16 operand fragment from a row-major matrix (global or LDS).
// ISA 7.12.2 (16-bit A-matrix 16x32): lane L -> row L%16; VGPR0..3 hold
// K = (L/16)*8 + 0..7, VGPR4..7 hold K = 16 + (L/16)*8 + 0..7.
__device__ __forceinline__ v16bf load_frag16(const __bf16* base, int stride, int k0) {
  const int lane = threadIdx.x & 31;
  const int row  = lane & 15;
  const int half = lane >> 4;
  const __bf16* p = base + row * stride + k0 + half * 8;
  union { v16bf v; uint4 u[2]; } f;
  f.u[0] = *(const uint4*)(p);
  f.u[1] = *(const uint4*)(p + 16);
  return f.v;
}

// ---------------- f32 -> bf16 conversion ----------------
__global__ __launch_bounds__(256) void cvt4_kernel(const float* __restrict__ in,
                                                   __bf16* __restrict__ out, int n4) {
  int i = blockIdx.x * blockDim.x + threadIdx.x;
  if (i >= n4) return;
  float4 f = ((const float4*)in)[i];
  v4bf r;
  r.x = (__bf16)f.x; r.y = (__bf16)f.y; r.z = (__bf16)f.z; r.w = (__bf16)f.w;
  ((v4bf*)out)[i] = r;
}

// ---------------- QKV projection GEMM -------------------
// Y[s,f] = sum_e X[s,e] * Wqkv[f,e]; the 64x32 weight tile is async-staged in
// LDS once per block per k-step (double buffered) and shared by all 8 waves.
// Epilogue scatters into q [H,S,Dk] (pre-scaled by 1/sqrt(Dk)), k [H,S,Dk],
// vT [H,Dk,S].
__global__ __launch_bounds__(256) void qkv_gemm_kernel(const __bf16* __restrict__ X,
                                                       const __bf16* __restrict__ W,
                                                       __bf16* __restrict__ q,
                                                       __bf16* __restrict__ k,
                                                       __bf16* __restrict__ vT) {
  __shared__ __align__(16) __bf16 bt[2][64][32];
  const int tid  = threadIdx.x;
  const int wave = tid >> 5;
  const int lane = tid & 31;
  const int m0 = blockIdx.x * 128 + wave * 16;   // sequence rows
  const int n0 = blockIdx.y * 64;                // output features

  auto stage = [&](int buf, int k0) {            // 1 async b128 per thread
    int r = tid >> 2, off = (tid & 3) * 8;
    async_copy16(W + (n0 + r) * EMBED + k0 + off, &bt[buf][r][off]);
  };

  stage(0, 0);
  v8f acc[4] = {};
  for (int i = 0; i < EMBED / 32; ++i) {
    const int k0  = i * 32;
    const int buf = i & 1;
    if (i + 1 < EMBED / 32) { stage(buf ^ 1, k0 + 32); async_wait<1>(); }
    else                    { async_wait<0>(); }
    __syncthreads();                              // tile[buf] visible to all waves
    v16bf a = load_frag16(X + m0 * EMBED, EMBED, k0);
#pragma unroll
    for (int t = 0; t < 4; ++t)
      acc[t] = wmma_bf16(a, load_frag16(&bt[buf][t * 16][0], 32, 0), acc[t]);
    __syncthreads();                              // reads done before restaging
  }

  const int half = lane >> 4, nn = lane & 15;
#pragma unroll
  for (int t = 0; t < 4; ++t) {
#pragma unroll
    for (int j = 0; j < 8; ++j) {
      int s = m0 + j + 8 * half;
      int f = n0 + t * 16 + nn;
      float v = acc[t][j];
      int which = f / EMBED;
      int r = f % EMBED;
      int h = r >> 6;
      int d = r & 63;
      if (which == 0)      q[(h * SEQ + s) * DK + d]  = (__bf16)(v * 0.125f);
      else if (which == 1) k[(h * SEQ + s) * DK + d]  = (__bf16)v;
      else                 vT[(h * DK + d) * SEQ + s] = (__bf16)v;
    }
  }
}

// ---------------- Flash attention -----------------------
// One workgroup = 1 head x 64 query rows (4 waves x 16 rows). K (32x64) and
// Vt (64x32) key-block tiles are async-staged into double-buffered LDS and
// shared by the 4 waves; P bounces through LDS to re-layout C/D -> A fragment.
__global__ __launch_bounds__(128) void attn_kernel(const __bf16* __restrict__ q,
                                                   const __bf16* __restrict__ kk,
                                                   const __bf16* __restrict__ vT,
                                                   __bf16* __restrict__ o) {
  __shared__ __align__(16) __bf16 kt[2][32][64];
  __shared__ __align__(16) __bf16 vt[2][64][32];
  __shared__ __align__(16) __bf16 p_lds[4][16][32];
  const int tid  = threadIdx.x;
  const int wave = tid >> 5;
  const int lane = tid & 31;
  const int h  = blockIdx.y;
  const int m0 = blockIdx.x * 64 + wave * 16;
  const int half = lane >> 4, nn = lane & 15;

  const __bf16* Q = q  + (h * SEQ + m0) * DK;
  const __bf16* K = kk + h * SEQ * DK;
  const __bf16* V = vT + h * DK * SEQ;

  auto stage = [&](int buf, int t0) {            // 4 async b128 per thread
#pragma unroll
    for (int c = tid; c < 256; c += 128) {
      { int r = c >> 3, off = (c & 7) * 8;       // K tile: [32][64]
        async_copy16(K + (t0 + r) * DK + off, &kt[buf][r][off]); }
      { int r = c >> 2, off = (c & 3) * 8;       // Vt tile: [64][32]
        async_copy16(V + r * SEQ + t0 + off, &vt[buf][r][off]); }
    }
  };

  const v16bf qa0 = load_frag16(Q, DK, 0);    // d = 0..31  (Q pre-scaled)
  const v16bf qa1 = load_frag16(Q, DK, 32);   // d = 32..63

  v8f acc[4] = {};
  float mrun[8], lrun[8];
#pragma unroll
  for (int j = 0; j < 8; ++j) { mrun[j] = -3.0e38f; lrun[j] = 0.0f; }

  stage(0, 0);
  for (int i = 0; i < SEQ / 32; ++i) {
    const int t0  = i * 32;
    const int buf = i & 1;
    if (i + 1 < SEQ / 32) { stage(buf ^ 1, t0 + 32); async_wait<4>(); }
    else                  { async_wait<0>(); }
    __syncthreads();                            // K/V tile[buf] ready

    // scores: two 16x16 tiles covering keys [t0, t0+32)
    v8f s0 = {}, s1 = {};
    s0 = wmma_bf16(qa0, load_frag16(&kt[buf][0][0],  64, 0),  s0);
    s0 = wmma_bf16(qa1, load_frag16(&kt[buf][0][0],  64, 32), s0);
    s1 = wmma_bf16(qa0, load_frag16(&kt[buf][16][0], 64, 0),  s1);
    s1 = wmma_bf16(qa1, load_frag16(&kt[buf][16][0], 64, 32), s1);

    float alpha[8];
#pragma unroll
    for (int j = 0; j < 8; ++j) {
      // row r = j + 8*(lane/16); its 16 values live in one 16-lane half
      float mx = fmaxf(s0[j], s1[j]);
      mx = fmaxf(mx, __shfl_xor(mx, 1, 32));
      mx = fmaxf(mx, __shfl_xor(mx, 2, 32));
      mx = fmaxf(mx, __shfl_xor(mx, 4, 32));
      mx = fmaxf(mx, __shfl_xor(mx, 8, 32));
      float mnew = fmaxf(mrun[j], mx);
      float a  = __expf(mrun[j] - mnew);
      float p0 = __expf(s0[j] - mnew);
      float p1 = __expf(s1[j] - mnew);
      float ps = p0 + p1;
      ps += __shfl_xor(ps, 1, 32);
      ps += __shfl_xor(ps, 2, 32);
      ps += __shfl_xor(ps, 4, 32);
      ps += __shfl_xor(ps, 8, 32);
      lrun[j] = lrun[j] * a + ps;
      mrun[j] = mnew;
      alpha[j] = a;
      int r = j + 8 * half;
      p_lds[wave][r][nn]      = (__bf16)p0;   // C/D layout -> row-major LDS
      p_lds[wave][r][nn + 16] = (__bf16)p1;
    }
#pragma unroll
    for (int t = 0; t < 4; ++t)
#pragma unroll
      for (int j = 0; j < 8; ++j) acc[t][j] *= alpha[j];

    __syncthreads();                            // p_lds writes visible
    v16bf pa = load_frag16(&p_lds[wave][0][0], 32, 0);   // 16x32 A fragment

    // O += P (16x32) * V (32x64) from the staged Vt tile
#pragma unroll
    for (int t = 0; t < 4; ++t)
      acc[t] = wmma_bf16(pa, load_frag16(&vt[buf][t * 16][0], 32, 0), acc[t]);

    __syncthreads();                            // all tile reads done
  }

  float inv[8];
#pragma unroll
  for (int j = 0; j < 8; ++j) inv[j] = 1.0f / lrun[j];
#pragma unroll
  for (int t = 0; t < 4; ++t) {
#pragma unroll
    for (int j = 0; j < 8; ++j) {
      int s = m0 + j + 8 * half;
      int d = t * 16 + nn;
      o[s * EMBED + h * DK + d] = (__bf16)(acc[t][j] * inv[j]);
    }
  }
}

// ---------------- Output projection GEMM ----------------
__global__ __launch_bounds__(256) void out_gemm_kernel(const __bf16* __restrict__ A,
                                                       const __bf16* __restrict__ W,
                                                       const float* __restrict__ bias,
                                                       float* __restrict__ out) {
  __shared__ __align__(16) __bf16 bt[2][64][32];
  const int tid  = threadIdx.x;
  const int wave = tid >> 5;
  const int lane = tid & 31;
  const int m0 = blockIdx.x * 128 + wave * 16;
  const int n0 = blockIdx.y * 64;

  auto stage = [&](int buf, int k0) {
    int r = tid >> 2, off = (tid & 3) * 8;
    async_copy16(W + (n0 + r) * EMBED + k0 + off, &bt[buf][r][off]);
  };

  stage(0, 0);
  v8f acc[4] = {};
  for (int i = 0; i < EMBED / 32; ++i) {
    const int k0  = i * 32;
    const int buf = i & 1;
    if (i + 1 < EMBED / 32) { stage(buf ^ 1, k0 + 32); async_wait<1>(); }
    else                    { async_wait<0>(); }
    __syncthreads();
    v16bf a = load_frag16(A + m0 * EMBED, EMBED, k0);
#pragma unroll
    for (int t = 0; t < 4; ++t)
      acc[t] = wmma_bf16(a, load_frag16(&bt[buf][t * 16][0], 32, 0), acc[t]);
    __syncthreads();
  }

  const int half = lane >> 4, nn = lane & 15;
#pragma unroll
  for (int t = 0; t < 4; ++t) {
#pragma unroll
    for (int j = 0; j < 8; ++j) {
      int s = m0 + j + 8 * half;
      int f = n0 + t * 16 + nn;
      out[s * EMBED + f] = acc[t][j] + bias[f];
    }
  }
}

// ---------------- host launcher -------------------------
extern "C" void kernel_launch(void* const* d_in, const int* in_sizes, int n_in,
                              void* d_out, int out_size, void* d_ws, size_t ws_size,
                              hipStream_t stream) {
  const float* x     = (const float*)d_in[0];   // [1,4096,768]
  const float* w_qkv = (const float*)d_in[1];   // [2304,768]
  const float* w_out = (const float*)d_in[2];   // [768,768]
  const float* b_out = (const float*)d_in[3];   // [768]
  float* out = (float*)d_out;

  char* ws = (char*)d_ws;
  // bf16 workspace layout (bytes, 16B-aligned):
  __bf16* xb    = (__bf16*)(ws + 0);            // 4096*768      = 6291456 B
  __bf16* wqkvb = (__bf16*)(ws + 6291456);      // 2304*768      = 3538944 B
  __bf16* woutb = (__bf16*)(ws + 9830400);      // 768*768       = 1179648 B
  __bf16* qb    = (__bf16*)(ws + 11010048);     // [H,S,Dk]      = 6291456 B
  __bf16* kb    = (__bf16*)(ws + 17301504);     // [H,S,Dk]      = 6291456 B
  __bf16* vtb   = (__bf16*)(ws + 23592960);     // [H,Dk,S]      = 6291456 B
  __bf16* ab    = (__bf16*)(ws + 29884416);     // [S,E] attnout = 6291456 B
                                                // total ~34.5 MB

  cvt4_kernel<<<dim3((SEQ * EMBED / 4 + 255) / 256), dim3(256), 0, stream>>>(
      x, xb, SEQ * EMBED / 4);
  cvt4_kernel<<<dim3((QKVD * EMBED / 4 + 255) / 256), dim3(256), 0, stream>>>(
      w_qkv, wqkvb, QKVD * EMBED / 4);
  cvt4_kernel<<<dim3((EMBED * EMBED / 4 + 255) / 256), dim3(256), 0, stream>>>(
      w_out, woutb, EMBED * EMBED / 4);

  qkv_gemm_kernel<<<dim3(SEQ / 128, QKVD / 64), dim3(256), 0, stream>>>(
      xb, wqkvb, qb, kb, vtb);

  attn_kernel<<<dim3(SEQ / 64, NHEAD), dim3(128), 0, stream>>>(qb, kb, vtb, ab);

  out_gemm_kernel<<<dim3(SEQ / 128, EMBED / 64), dim3(256), 0, stream>>>(
      ab, woutb, b_out, out);
}
